// GraphEncoder_14886356648544
// MI455X (gfx1250) — matrix operand order repaired and checked by reading, not compiled, not run on hardware
//
#include <hip/hip_runtime.h>

#define N_NODES 100000
#define N_EDGES 1000000
#define D       128

typedef __attribute__((ext_vector_type(16))) _Float16 v16h;
typedef __attribute__((ext_vector_type(8)))  float    v8f;

// ---------------------------------------------------------------------------
// hardware f32 atomic add (global_atomic_add_f32)
// ---------------------------------------------------------------------------
__device__ __forceinline__ void atomic_fadd(float* p, float v) {
    unsafeAtomicAdd(p, v);
}

// ---------------------------------------------------------------------------
// zero a float buffer
// ---------------------------------------------------------------------------
__global__ void zero_f_kernel(float* __restrict__ p, int n) {
    int i = blockIdx.x * blockDim.x + threadIdx.x;
    if (i < n) p[i] = 0.0f;
}

// ---------------------------------------------------------------------------
// deg[dst] += 1 per edge (in-degree)
// ---------------------------------------------------------------------------
__global__ void degree_kernel(const long long* __restrict__ dst,
                              float* __restrict__ deg, int nE) {
    int e = blockIdx.x * blockDim.x + threadIdx.x;
    if (e < nE) atomic_fadd(&deg[(int)dst[e]], 1.0f);
}

// ---------------------------------------------------------------------------
// dinv[i] = rsqrt(1 + deg[i])   (in place)
// ---------------------------------------------------------------------------
__global__ void dinv_kernel(float* __restrict__ d, int n) {
    int i = blockIdx.x * blockDim.x + threadIdx.x;
    if (i < n) d[i] = __frsqrt_rn(1.0f + d[i]);
}

// ---------------------------------------------------------------------------
// Wt[n][k] = (f16) W[k][n]   for both layer weights (128x128 each)
// ---------------------------------------------------------------------------
__global__ void prep_weights_kernel(const float* __restrict__ W1,
                                    const float* __restrict__ W2,
                                    _Float16* __restrict__ Wt1,
                                    _Float16* __restrict__ Wt2) {
    int i = blockIdx.x * blockDim.x + threadIdx.x;   // 0..16383
    if (i >= D * D) return;
    int k = i >> 7, n = i & (D - 1);
    Wt1[(size_t)n * D + k] = (_Float16)W1[i];
    Wt2[(size_t)n * D + k] = (_Float16)W2[i];
}

// ---------------------------------------------------------------------------
// WMMA GEMM: H = A(f32,[M,128]) @ W  using f16 inputs / f32 accum.
// Also writes AGG = H * dinv[row]^2  (self-loop term of GCN aggregation).
// One wave -> one 16x16 tile; block = 8 waves = all 8 N-tiles of one M-tile.
// ---------------------------------------------------------------------------
__global__ __launch_bounds__(256) void gemm_gcn_kernel(
        const float*    __restrict__ A,     // [M,128] row-major f32
        const _Float16* __restrict__ Wt,    // [128,128] f16, Wt[n][k]=W[k][n]
        const float*    __restrict__ dinv,  // [M]
        float*          __restrict__ H,     // [M,128] out: A@W
        float*          __restrict__ AGG,   // [M,128] out: H*dinv^2
        int M) {
    const int lane  = threadIdx.x & 31;
    const int ntile = threadIdx.x >> 5;        // 0..7
    const int row0  = blockIdx.x * 16;
    if (row0 >= M) return;
    const int half = lane >> 4;                // 0 or 1
    const int l16  = lane & 15;

    const float*    arow = A  + (size_t)(row0 + l16) * D;          // A row M=l16
    const _Float16* brow = Wt + (size_t)(ntile * 16 + l16) * D;    // B col N=l16

    v8f c = {0.f, 0.f, 0.f, 0.f, 0.f, 0.f, 0.f, 0.f};

#pragma unroll
    for (int ks = 0; ks < 4; ++ks) {           // K = 128 in steps of 32
        const int kb = ks * 32;
        // --- A fragment: lane holds K = kb + half*16 + {0..15}
        const int ka = kb + half * 16;
        float4 f0 = *(const float4*)(arow + ka);
        float4 f1 = *(const float4*)(arow + ka + 4);
        float4 f2 = *(const float4*)(arow + ka + 8);
        float4 f3 = *(const float4*)(arow + ka + 12);
        v16h a;
        a[0]  = (_Float16)f0.x; a[1]  = (_Float16)f0.y;
        a[2]  = (_Float16)f0.z; a[3]  = (_Float16)f0.w;
        a[4]  = (_Float16)f1.x; a[5]  = (_Float16)f1.y;
        a[6]  = (_Float16)f1.z; a[7]  = (_Float16)f1.w;
        a[8]  = (_Float16)f2.x; a[9]  = (_Float16)f2.y;
        a[10] = (_Float16)f2.z; a[11] = (_Float16)f2.w;
        a[12] = (_Float16)f3.x; a[13] = (_Float16)f3.y;
        a[14] = (_Float16)f3.z; a[15] = (_Float16)f3.w;
        // --- B fragment: lane holds K = kb + half*16 + e (contiguous in Wt row)
        v16h b = *(const v16h*)(brow + kb + half * 16);
        // D = A*B + C
        c = __builtin_amdgcn_wmma_f32_16x16x32_f16(
                /*neg_a=*/false, a, /*neg_b=*/false, b,
                /*c_mod=*/(short)0, c, /*reuse_a=*/false, /*reuse_b=*/false);
    }

    // Epilogue: VGPR g -> row = row0 + g + half*8, col = ntile*16 + l16
#pragma unroll
    for (int g = 0; g < 8; ++g) {
        const int r   = row0 + g + half * 8;
        const int col = ntile * 16 + l16;
        const float v  = c[g];
        const float di = dinv[r];
        H  [(size_t)r * D + col] = v;
        AGG[(size_t)r * D + col] = v * di * di;
    }
}

// ---------------------------------------------------------------------------
// Edge scatter: AGG[dst] += H[src] * (dinv[src]*dinv[dst]).
// One wave per edge; each lane handles 4 features (float4 gather + 4 atomics).
// ---------------------------------------------------------------------------
__global__ __launch_bounds__(256) void scatter_kernel(
        const long long* __restrict__ src,
        const long long* __restrict__ dst,
        const float*     __restrict__ H,
        const float*     __restrict__ dinv,
        float*           __restrict__ AGG,
        int nE) {
    const int e = blockIdx.x * 8 + (threadIdx.x >> 5);
    if (e >= nE) return;
    const int lane = threadIdx.x & 31;
    const int s = (int)src[e];
    const int d = (int)dst[e];
    const float coef = dinv[s] * dinv[d];
    const float4 v = *(const float4*)(H + (size_t)s * D + lane * 4);
    float* out = AGG + (size_t)d * D + lane * 4;
    atomic_fadd(out + 0, v.x * coef);
    atomic_fadd(out + 1, v.y * coef);
    atomic_fadd(out + 2, v.z * coef);
    atomic_fadd(out + 3, v.w * coef);
}

// ---------------------------------------------------------------------------
// X2 = relu(AGG + b)   (layer-1 activation, feeds layer-2 GEMM)
// ---------------------------------------------------------------------------
__global__ void bias_relu_kernel(const float* __restrict__ agg,
                                 const float* __restrict__ b,
                                 float* __restrict__ out, int total) {
    int i = blockIdx.x * blockDim.x + threadIdx.x;
    if (i < total) {
        float v = agg[i] + b[i & (D - 1)];
        out[i] = v > 0.0f ? v : 0.0f;
    }
}

// ---------------------------------------------------------------------------
// out[d] = b2[d]  (init; d_out is poisoned by harness)
// ---------------------------------------------------------------------------
__global__ void out_init_kernel(const float* __restrict__ b2,
                                float* __restrict__ out) {
    int d = threadIdx.x;
    if (d < D) out[d] = b2[d];
}

// ---------------------------------------------------------------------------
// out[d] += (1/N) * sum_rows AGG2[r][d]
// ---------------------------------------------------------------------------
__global__ __launch_bounds__(128) void reduce_mean_kernel(
        const float* __restrict__ agg2, float* __restrict__ out, int n) {
    const int d = threadIdx.x;          // feature 0..127
    float s = 0.0f;
    for (int r = blockIdx.x; r < n; r += gridDim.x)
        s += agg2[(size_t)r * D + d];
    atomic_fadd(&out[d], s * (1.0f / (float)N_NODES));
}

// ---------------------------------------------------------------------------
extern "C" void kernel_launch(void* const* d_in, const int* in_sizes, int n_in,
                              void* d_out, int out_size, void* d_ws, size_t ws_size,
                              hipStream_t stream) {
    const float*     x    = (const float*)d_in[0];
    const long long* eidx = (const long long*)d_in[1];   // int64 [2, E]
    const float*     W1   = (const float*)d_in[2];
    const float*     b1   = (const float*)d_in[3];
    const float*     W2   = (const float*)d_in[4];
    const float*     b2   = (const float*)d_in[5];
    float* out = (float*)d_out;

    const long long* src = eidx;
    const long long* dst = eidx + N_EDGES;

    // workspace layout (all offsets 512B-aligned)
    const size_t SZ = (size_t)N_NODES * D * sizeof(float);   // 51,200,000 B
    char* ws = (char*)d_ws;
    float*    bufA = (float*)(ws);              // H1, then X2
    float*    bufB = (float*)(ws + SZ);         // AGG1, then H2
    float*    bufC = (float*)(ws + 2 * SZ);     // AGG2
    float*    dinv = (float*)(ws + 3 * SZ);     // deg -> dinv, N floats
    _Float16* Wt1  = (_Float16*)(ws + 3 * SZ + (1 << 20));
    _Float16* Wt2  = (_Float16*)(ws + 3 * SZ + (1 << 20) + D * D * sizeof(_Float16));

    const int nElem   = N_NODES * D;
    const int mtiles  = N_NODES / 16;           // 6250 (N_NODES % 16 == 0)

    // 1) degrees -> dinv
    zero_f_kernel<<<(N_NODES + 255) / 256, 256, 0, stream>>>(dinv, N_NODES);
    degree_kernel<<<(N_EDGES + 255) / 256, 256, 0, stream>>>(dst, dinv, N_EDGES);
    dinv_kernel<<<(N_NODES + 255) / 256, 256, 0, stream>>>(dinv, N_NODES);

    // 2) f16-transposed weights
    prep_weights_kernel<<<(D * D + 255) / 256, 256, 0, stream>>>(W1, W2, Wt1, Wt2);

    // 3) layer 1: H1 = x@W1 (+ self-loop init), edge scatter, relu(+b1)
    gemm_gcn_kernel<<<mtiles, 256, 0, stream>>>(x, Wt1, dinv, bufA, bufB, N_NODES);
    scatter_kernel<<<(N_EDGES + 7) / 8, 256, 0, stream>>>(src, dst, bufA, dinv, bufB, N_EDGES);
    bias_relu_kernel<<<(nElem + 255) / 256, 256, 0, stream>>>(bufB, b1, bufA, nElem);

    // 4) layer 2: H2 = X2@W2 (+ self-loop init), edge scatter
    gemm_gcn_kernel<<<mtiles, 256, 0, stream>>>(bufA, Wt2, dinv, bufB, bufC, N_NODES);
    scatter_kernel<<<(N_EDGES + 7) / 8, 256, 0, stream>>>(src, dst, bufB, dinv, bufC, N_EDGES);

    // 5) global mean + b2
    out_init_kernel<<<1, 128, 0, stream>>>(b2, out);
    reduce_mean_kernel<<<512, 128, 0, stream>>>(bufC, out, N_NODES);
}
